// BertSelfAttention_55748675502833
// MI455X (gfx1250) — matrix operand order repaired and checked by reading, not compiled, not run on hardware
//
#include <hip/hip_runtime.h>

// ---------------------------------------------------------------------------
// BERT self-attention, fused flash-style, bf16 WMMA (gfx1250 / CDNA5, wave32)
// B=4, S=2048, D=1024, H=16, d=64
// ---------------------------------------------------------------------------

typedef __attribute__((ext_vector_type(16))) __bf16 v16bf;
typedef __attribute__((ext_vector_type(8)))  __bf16 v8bf;
typedef __attribute__((ext_vector_type(8)))  float  v8f;

#define WMMA_BF16(A, B, C) \
  __builtin_amdgcn_wmma_f32_16x16x32_bf16(false, (A), false, (B), (short)0, (C), false, false)

static constexpr int B_  = 4;
static constexpr int S_  = 2048;
static constexpr int D_  = 1024;
static constexpr int H_  = 16;
static constexpr int HD_ = 64;

// ---------------------------------------------------------------------------
// fp32 -> bf16 convert (vectorized x4; all sizes here are multiples of 1024)
// ---------------------------------------------------------------------------
__global__ __launch_bounds__(256) void cvt_f32_bf16(const float* __restrict__ src,
                                                    __bf16* __restrict__ dst, int n4) {
  int i = (blockIdx.x * blockDim.x + threadIdx.x);
  if (i < n4) {
    float4 f = ((const float4*)src)[i];
    __bf16* d = dst + (size_t)i * 4;
    d[0] = (__bf16)f.x; d[1] = (__bf16)f.y; d[2] = (__bf16)f.z; d[3] = (__bf16)f.w;
  }
}

// ---------------------------------------------------------------------------
// Fused QKV projection: Y = X @ W^T + bias, scale folded into Q.
// blockIdx.z selects q/k/v. Block = 8 waves; wave tile 32(M) x 64(N)
// (2x4 WMMA tiles), block tile 128 x 128.
// Q,K out: [B,H,S,64] bf16 ; V out transposed: [B,H,64,S] bf16.
// ---------------------------------------------------------------------------
__global__ __launch_bounds__(256)
void qkv_gemm(const __bf16* __restrict__ X,
              const __bf16* __restrict__ Wq, const __bf16* __restrict__ Wk,
              const __bf16* __restrict__ Wv,
              const float* __restrict__ bq, const float* __restrict__ bk,
              const float* __restrict__ bv,
              __bf16* __restrict__ Qo, __bf16* __restrict__ Ko, __bf16* __restrict__ Vt)
{
  const int z = blockIdx.z;
  const __bf16* W   = (z == 0) ? Wq : (z == 1) ? Wk : Wv;
  const float* bias = (z == 0) ? bq : (z == 1) ? bk : bv;

  const int lane = threadIdx.x & 31;
  const int wave = threadIdx.x >> 5;
  const int col  = lane & 15;      // C-layout column / A row / B column
  const int half = lane >> 4;      // lane half selects K-chunk / C row block

  const int m_base = blockIdx.x * 128 + (wave >> 1) * 32;
  const int n_base = blockIdx.y * 128 + (wave & 1) * 64;

  v8f acc[2][4] = {};

  const __bf16* xp0 = X + (size_t)(m_base + col) * D_ + half * 16;
  const __bf16* xp1 = X + (size_t)(m_base + 16 + col) * D_ + half * 16;

  for (int k0 = 0; k0 < D_; k0 += 32) {
    v16bf a[2], bb[4];
    {
      v8bf lo0 = *(const v8bf*)(xp0 + k0);
      v8bf hi0 = *(const v8bf*)(xp0 + k0 + 8);
      v8bf lo1 = *(const v8bf*)(xp1 + k0);
      v8bf hi1 = *(const v8bf*)(xp1 + k0 + 8);
#pragma unroll
      for (int e = 0; e < 8; ++e) {
        a[0][e] = lo0[e]; a[0][8 + e] = hi0[e];
        a[1][e] = lo1[e]; a[1][8 + e] = hi1[e];
      }
    }
#pragma unroll
    for (int ni = 0; ni < 4; ++ni) {
      const __bf16* p = W + (size_t)(n_base + ni * 16 + col) * D_ + k0 + half * 16;
      bb[ni] = *(const v16bf*)p;
    }
#pragma unroll
    for (int mi = 0; mi < 2; ++mi)
#pragma unroll
      for (int ni = 0; ni < 4; ++ni)
        acc[mi][ni] = WMMA_BF16(a[mi], bb[ni], acc[mi][ni]);
  }

  // Epilogue: bias (+scale for Q). Uniform branch hoisted outside the loops.
  if (z == 2) {
    // V: store transposed [B,H,d,S] so the PV B-fragment read is contiguous.
#pragma unroll
    for (int ni = 0; ni < 4; ++ni) {
      const int n  = n_base + ni * 16 + col;
      const int h  = n >> 6;
      const int dd = n & 63;
      const float bval = bias[n];
#pragma unroll
      for (int mi = 0; mi < 2; ++mi)
#pragma unroll
        for (int v = 0; v < 8; ++v) {
          const int m  = m_base + mi * 16 + half * 8 + v;
          const int bi = m >> 11;
          const int s  = m & (S_ - 1);
          Vt[(((size_t)(bi * H_ + h)) * HD_ + dd) * S_ + s] = (__bf16)(acc[mi][ni][v] + bval);
        }
    }
  } else {
    __bf16* out = (z == 0) ? Qo : Ko;
    const float scale = (z == 0) ? 0.125f : 1.0f;   // 1/sqrt(64) folded into Q
#pragma unroll
    for (int ni = 0; ni < 4; ++ni) {
      const int n  = n_base + ni * 16 + col;
      const int h  = n >> 6;
      const int dd = n & 63;
      const float bval = bias[n];
#pragma unroll
      for (int mi = 0; mi < 2; ++mi)
#pragma unroll
        for (int v = 0; v < 8; ++v) {
          const int m  = m_base + mi * 16 + half * 8 + v;
          const int bi = m >> 11;
          const int s  = m & (S_ - 1);
          out[(((size_t)(bi * H_ + h)) * S_ + s) * HD_ + dd] =
              (__bf16)((acc[mi][ni][v] + bval) * scale);
        }
    }
  }
}

// ---------------------------------------------------------------------------
// Flash attention: ctx = softmax(Q K^T * scale + mask) V, out [B,S,H*d] f32.
// Block = 8 waves, 128 queries; each wave owns 16 query rows, streams keys
// in 32-wide tiles with online softmax. P routed through wave-private LDS to
// convert C-layout f32 -> A-layout bf16.
// ---------------------------------------------------------------------------
__global__ __launch_bounds__(256)
void flash_attn(const __bf16* __restrict__ Q, const __bf16* __restrict__ K,
                const __bf16* __restrict__ Vt, const float* __restrict__ mask,
                float* __restrict__ out)
{
  const int lane = threadIdx.x & 31;
  const int wave = threadIdx.x >> 5;
  const int col  = lane & 15;
  const int half = lane >> 4;
  const int h  = blockIdx.y;
  const int b  = blockIdx.z;
  const int bh = b * H_ + h;
  const int q0 = blockIdx.x * 128 + wave * 16;

  __shared__ __bf16 pLds[8 * 16 * 32];
  __bf16* myP = pLds + wave * 512;     // wave-private 16x32 staging tile

  // Q A-fragments (dd 0..31 and 32..63), resident for the whole kernel.
  v16bf qa[2];
  {
    const __bf16* qp = Q + ((size_t)bh * S_ + q0 + col) * HD_;
#pragma unroll
    for (int kk = 0; kk < 2; ++kk) {
      v8bf lo = *(const v8bf*)(qp + kk * 32 + half * 16);
      v8bf hi = *(const v8bf*)(qp + kk * 32 + half * 16 + 8);
#pragma unroll
      for (int e = 0; e < 8; ++e) { qa[kk][e] = lo[e]; qa[kk][8 + e] = hi[e]; }
    }
  }

  const __bf16* kbase = K + ((size_t)bh * S_ + col) * HD_ + half * 16;
  const __bf16* vbase = Vt + ((size_t)bh * HD_ + col) * S_ + half * 16;

  v8f o[4] = {};                       // 16x64 f32 output accumulator
  float m_i[8], l_i[8];
#pragma unroll
  for (int v = 0; v < 8; ++v) { m_i[v] = -1e30f; l_i[v] = 0.0f; }

  for (int kt = 0; kt < S_; kt += 32) {
    // ---- scores: S = Q K^T (Q pre-scaled), two 16x16 tiles ----
    v8f sc[2] = {};
#pragma unroll
    for (int t = 0; t < 2; ++t) {
      const __bf16* kp = kbase + (size_t)(kt + t * 16) * HD_;
#pragma unroll
      for (int kk = 0; kk < 2; ++kk) {
        v16bf kb = *(const v16bf*)(kp + kk * 32);
        sc[t] = WMMA_BF16(qa[kk], kb, sc[t]);
      }
    }
    // prefetch next key tile (linear stream, exact address)
    if (kt + 32 < S_) __builtin_prefetch(kbase + (size_t)(kt + 32) * HD_, 0, 0);
    // ---- additive attention mask ----
#pragma unroll
    for (int t = 0; t < 2; ++t) {
      const float mval = mask[b * S_ + kt + t * 16 + col];
#pragma unroll
      for (int v = 0; v < 8; ++v) sc[t][v] += mval;
    }
    // ---- online softmax: row max over this 32-key tile ----
    float alpha[8];
#pragma unroll
    for (int v = 0; v < 8; ++v) {
      float x = fmaxf(sc[0][v], sc[1][v]);
      x = fmaxf(x, __shfl_xor(x, 1, 16));
      x = fmaxf(x, __shfl_xor(x, 2, 16));
      x = fmaxf(x, __shfl_xor(x, 4, 16));
      x = fmaxf(x, __shfl_xor(x, 8, 16));
      const float mn = fmaxf(m_i[v], x);
      alpha[v] = __expf(m_i[v] - mn);
      m_i[v] = mn;
    }
    // ---- P = exp(S - m), row sums, l update ----
#pragma unroll
    for (int v = 0; v < 8; ++v) {
      sc[0][v] = __expf(sc[0][v] - m_i[v]);
      sc[1][v] = __expf(sc[1][v] - m_i[v]);
      float s2 = sc[0][v] + sc[1][v];
      s2 += __shfl_xor(s2, 1, 16);
      s2 += __shfl_xor(s2, 2, 16);
      s2 += __shfl_xor(s2, 4, 16);
      s2 += __shfl_xor(s2, 8, 16);
      l_i[v] = l_i[v] * alpha[v] + s2;
    }
    // ---- C-layout -> A-layout transpose of P via wave-private LDS ----
#pragma unroll
    for (int t = 0; t < 2; ++t)
#pragma unroll
      for (int v = 0; v < 8; ++v)
        myP[(half * 8 + v) * 32 + t * 16 + col] = (__bf16)sc[t][v];
    v16bf pa;
    {
      const __bf16* pp = myP + col * 32;
      v8bf lo = *(const v8bf*)(pp + half * 16);
      v8bf hi = *(const v8bf*)(pp + half * 16 + 8);
#pragma unroll
      for (int e = 0; e < 8; ++e) { pa[e] = lo[e]; pa[8 + e] = hi[e]; }
    }
    // ---- O = O*alpha + P @ V (V transposed: contiguous along keys) ----
#pragma unroll
    for (int nd = 0; nd < 4; ++nd) {
#pragma unroll
      for (int v = 0; v < 8; ++v) o[nd][v] *= alpha[v];
      v16bf vb = *(const v16bf*)(vbase + (size_t)nd * 16 * S_ + kt);
      o[nd] = WMMA_BF16(pa, vb, o[nd]);
    }
    if (kt + 32 < S_) __builtin_prefetch(vbase + kt + 32, 0, 0);
  }

  // ---- normalize and write [B,S,H*d] fp32 ----
  float invl[8];
#pragma unroll
  for (int v = 0; v < 8; ++v) invl[v] = 1.0f / l_i[v];
#pragma unroll
  for (int nd = 0; nd < 4; ++nd) {
#pragma unroll
    for (int v = 0; v < 8; ++v) {
      const int s  = q0 + half * 8 + v;
      const int dd = h * HD_ + nd * 16 + col;
      out[((size_t)(b * S_ + s)) * D_ + dd] = o[nd][v] * invl[v];
    }
  }
}

// ---------------------------------------------------------------------------
extern "C" void kernel_launch(void* const* d_in, const int* in_sizes, int n_in,
                              void* d_out, int out_size, void* d_ws, size_t ws_size,
                              hipStream_t stream) {
  (void)in_sizes; (void)n_in; (void)out_size; (void)ws_size;
  const float* hs   = (const float*)d_in[0];
  const float* mask = (const float*)d_in[1];
  const float* qw   = (const float*)d_in[2];
  const float* qb   = (const float*)d_in[3];
  const float* kw   = (const float*)d_in[4];
  const float* kb   = (const float*)d_in[5];
  const float* vw   = (const float*)d_in[6];
  const float* vb   = (const float*)d_in[7];
  float* out = (float*)d_out;

  char* ws = (char*)d_ws;
  size_t off = 0;
  auto wsAlloc = [&](size_t bytes) -> void* {
    void* p = ws + off;
    off = (off + bytes + 255) & ~(size_t)255;
    return p;
  };
  const size_t nX = (size_t)B_ * S_ * D_;       // 8.4M
  const size_t nW = (size_t)D_ * D_;            // 1.05M
  __bf16* Xbf  = (__bf16*)wsAlloc(nX * 2);
  __bf16* Wqbf = (__bf16*)wsAlloc(nW * 2);
  __bf16* Wkbf = (__bf16*)wsAlloc(nW * 2);
  __bf16* Wvbf = (__bf16*)wsAlloc(nW * 2);
  __bf16* Qbf  = (__bf16*)wsAlloc(nX * 2);      // [B,H,S,64]
  __bf16* Kbf  = (__bf16*)wsAlloc(nX * 2);      // [B,H,S,64]
  __bf16* Vtbf = (__bf16*)wsAlloc(nX * 2);      // [B,H,64,S]

  // 1) fp32 -> bf16 conversions
  {
    int n4 = (int)(nX / 4);
    cvt_f32_bf16<<<(n4 + 255) / 256, 256, 0, stream>>>(hs, Xbf, n4);
    int w4 = (int)(nW / 4);
    cvt_f32_bf16<<<(w4 + 255) / 256, 256, 0, stream>>>(qw, Wqbf, w4);
    cvt_f32_bf16<<<(w4 + 255) / 256, 256, 0, stream>>>(kw, Wkbf, w4);
    cvt_f32_bf16<<<(w4 + 255) / 256, 256, 0, stream>>>(vw, Wvbf, w4);
  }

  // 2) fused QKV projection (z = q/k/v), block tile 128x128
  qkv_gemm<<<dim3((B_ * S_) / 128, D_ / 128, 3), 256, 0, stream>>>(
      Xbf, Wqbf, Wkbf, Wvbf, qb, kb, vb, Qbf, Kbf, Vtbf);

  // 3) flash attention
  flash_attn<<<dim3(S_ / 128, H_, B_), 256, 0, stream>>>(Qbf, Kbf, Vtbf, mask, out);
}
